// GNNExpertCoupler_63213328662861
// MI455X (gfx1250) — matrix operand order repaired and checked by reading, not compiled, not run on hardware
//
#include <hip/hip_runtime.h>
#include <hip/hip_bf16.h>

// ---- problem constants (from reference) ----
// B=8, S=2048, E=8, D=512, L=2
#define NTOK   16384      // B*S
#define NE     8          // experts
#define DDIM   512
#define NROWS  (NTOK*NE)  // 131072 GEMM rows per layer
#define NTILES 32         // DDIM/16 column tiles of B

typedef _Float16 v16h __attribute__((ext_vector_type(16)));
typedef float    v8f  __attribute__((ext_vector_type(8)));

__device__ __forceinline__ float sigmoidf_(float x) { return 1.0f / (1.0f + expf(-x)); }
__device__ __forceinline__ float gelu_exact_(float x) {
    return 0.5f * x * (1.0f + erff(x * 0.70710678118654752f));
}

// ---- CDNA5 async global->LDS DMA (ASYNCcnt-tracked), per-lane 16B ----
__device__ __forceinline__ void async_copy16(void* lds_ptr, const void* gptr) {
    // low 32 bits of the generic LDS address are the workgroup-relative offset
    unsigned lds_off = (unsigned)(unsigned long long)lds_ptr;
    asm volatile("global_load_async_to_lds_b128 %0, %1, off"
                 :: "v"(lds_off), "v"(gptr)
                 : "memory");
}
__device__ __forceinline__ void wait_async0() {
    asm volatile("s_wait_asynccnt 0x0" ::: "memory");
}
// async loads complete in order: <=4 outstanding means all but the most
// recently issued tile (4 ops) have landed in LDS
__device__ __forceinline__ void wait_async4() {
    asm volatile("s_wait_asynccnt 0x4" ::: "memory");
}

// ---------------------------------------------------------------------------
// Weight pre-swizzle into WMMA B-fragment layout.
// B element (k, n) lives in block (k>>5, n>>4); within a block each lane
// holds 16 contiguous halves: lane = (n&15) + (k&16), half = (k&15).
// Block is 32 lanes * 16 halves = 512 halves = 1KB, contiguous; blocks are
// ct-major within a k-tile, so a 128-column B tile is one contiguous 8KB run.
// ---------------------------------------------------------------------------
__device__ __forceinline__ size_t bswz_idx(int k, int n) {
    return ((size_t)((k >> 5) * NTILES + (n >> 4)) << 9) +
           (size_t)(((n & 15) + (k & 16)) << 4) + (size_t)(k & 15);
}

__global__ __launch_bounds__(256) void build_b_layer_kernel(
    const float* __restrict__ Wn, const float* __restrict__ Ws,
    _Float16* __restrict__ Bswz) {
    const int l   = blockIdx.y;
    const int idx = blockIdx.x * 256 + threadIdx.x;   // 0 .. 1024*512-1
    const int k   = idx >> 9;                          // 0..1023
    const int n   = idx & 511;
    float v = (k < DDIM)
        ? Wn[((size_t)l * DDIM + k) * DDIM + n]
        : Ws[((size_t)l * DDIM + (k - DDIM)) * DDIM + n];
    Bswz[(size_t)l * (1024 * 512) + bswz_idx(k, n)] = (_Float16)v;
}

__global__ __launch_bounds__(256) void build_b_head_kernel(
    const float* __restrict__ Wc, _Float16* __restrict__ Bswz) {
    const int idx = blockIdx.x * 256 + threadIdx.x;   // 0 .. 512*512-1
    const int k   = idx >> 9;
    const int n   = idx & 511;
    Bswz[bswz_idx(k, n)] = (_Float16)Wc[(size_t)k * DDIM + n];
}

__global__ void build_bias_kernel(const float* __restrict__ bn,
                                  const float* __restrict__ bs,
                                  float* __restrict__ bias2) {
    const int l = blockIdx.x;
    const int n = threadIdx.x;       // 512 threads
    bias2[l * DDIM + n] = bn[l * DDIM + n] + bs[l * DDIM + n];
}

// ---------------------------------------------------------------------------
// Per-token gate + message aggregation. One block per token.
// Emits f16 A-matrix row block:  A[row, 0:512] = msg,  A[row, 512:1024] = f.
// ---------------------------------------------------------------------------
__global__ __launch_bounds__(256) void gate_msg_kernel(
    const float* __restrict__ f,      // (NTOK*NE*DDIM) fp32
    const float* __restrict__ adjl,   // (NE*NE) this layer
    const float* __restrict__ wm,     // (2*DDIM) this layer: [w1 | w2]
    const float* __restrict__ bm,     // scalar this layer
    _Float16* __restrict__ Aout)      // (NROWS x 1024) f16
{
    __shared__ float fs[NE * DDIM];       // 16 KB
    __shared__ float av[NE], bv[NE];
    __shared__ float gate[NE * NE];

    const int tid = threadIdx.x;
    const size_t tok = blockIdx.x;
    // async DMA the 16 KB token tile into LDS (consumed from LDS only)
    const float* fsrc = f + tok * (NE * DDIM);
    #pragma unroll
    for (int i = 0; i < 4; ++i) {
        int c = tid + i * 256;            // 1024 chunks of 16 B
        async_copy16(&fs[c * 4], fsrc + c * 4);
    }
    wait_async0();
    __syncthreads();

    const int w = tid >> 5, lane = tid & 31;
    // per-expert dot products: a = f.w1 , b = f.w2  (wave w owns expert w)
    float sa = 0.f, sb = 0.f;
    #pragma unroll
    for (int i = 0; i < DDIM / 32; ++i) {
        int d = lane + i * 32;
        float v = fs[w * DDIM + d];
        sa += v * wm[d];
        sb += v * wm[DDIM + d];
    }
    // wave32 butterfly reduction
    #pragma unroll
    for (int off = 16; off > 0; off >>= 1) {
        sa += __shfl_xor(sa, off, 32);
        sb += __shfl_xor(sb, off, 32);
    }
    if (lane == 0) { av[w] = sa; bv[w] = sb; }
    __syncthreads();

    if (tid < 64) {
        int i = tid >> 3, j = tid & 7;
        float adj = sigmoidf_(adjl[tid]);
        float g = adj * sigmoidf_(av[i] + bv[j] + bm[0]);
        gate[tid] = (i == j) ? 0.f : g;
    }
    __syncthreads();

    // msg[i,d] = sum_j gate[i][j] * f[j,d]; pack [msg | f] as f16
    _Float16* arow = Aout + tok * (size_t)(NE * 1024);
    float g[NE];
    #pragma unroll
    for (int j = 0; j < NE; ++j) g[j] = gate[w * NE + j];
    #pragma unroll
    for (int i = 0; i < DDIM / 32; ++i) {
        int d = lane + i * 32;
        float m = 0.f;
        #pragma unroll
        for (int j = 0; j < NE; ++j) m += g[j] * fs[j * DDIM + d];
        arow[(size_t)w * 1024 + d]        = (_Float16)m;
        arow[(size_t)w * 1024 + DDIM + d] = (_Float16)fs[w * DDIM + d];
    }
}

// ---------------------------------------------------------------------------
// WMMA GEMM: out[m,n] = epilogue( sum_k A[m,k]*B[k,n] + bias[n] )
//  - A: f16 row-major, lda halves
//  - B: pre-swizzled f16 fragment blocks (see bswz_idx), N fixed = 512
//  - 256 threads = 8 waves; block tile 128x128; wave = 16-row strip x 128 cols
//  - 3-stage LDS pipeline (48 KB) filled by ASYNC global->LDS DMA; the
//    in-order ASYNCcnt wait (<=4) only drains the tile issued one full
//    iteration earlier, so DMA latency is covered by a whole WMMA chain
//  - epilogue: +bias, optional exact GELU, optional +residual (fp32, ld=512)
// ---------------------------------------------------------------------------
__global__ __launch_bounds__(256) void gemm_kernel(
    const _Float16* __restrict__ A, int lda,
    const _Float16* __restrict__ B, int K,
    const float* __restrict__ bias,
    const float* __restrict__ res,
    float* __restrict__ out,
    int do_gelu)
{
    __shared__ _Float16 As[3][8 * 32 * 16];   // 3 x 8 KB
    __shared__ _Float16 Bs[3][8 * 512];       // 3 x 8 KB (8 fragment blocks)

    const int tid  = threadIdx.x;
    const int w    = tid >> 5;
    const int lane = tid & 31;
    const int mBase  = blockIdx.x * 128;
    const int nBase  = blockIdx.y * 128;
    const int ctBase = nBase >> 4;

    v8f zero = {};
    v8f acc[8];
    #pragma unroll
    for (int t = 0; t < 8; ++t) acc[t] = zero;

    auto load_tile = [&](int kt, int buf) {
        // A tile (128 rows x 32 halves): 16B global chunk -> 16B swizzled LDS slot
        #pragma unroll
        for (int cc = 0; cc < 2; ++cc) {
            int c = tid + cc * 256;          // 512 chunks of 8 halves
            int m = c >> 2;                  // 0..127
            int q = c & 3;                   // which 8-half group of the 32
            const void* g = (const void*)(A + (size_t)(mBase + m) * lda + kt * 32 + q * 8);
            async_copy16(&As[buf][((((m >> 4) * 32) + (m & 15) + ((q >> 1) << 4)) << 4)
                                  + ((q & 1) << 3)], g);
        }
        // B tile: one contiguous 8 KB run of pre-swizzled fragment blocks
        const _Float16* bsrc = B + (((size_t)kt * NTILES + ctBase) << 9);
        #pragma unroll
        for (int cc = 0; cc < 2; ++cc) {
            int c = tid + cc * 256;
            async_copy16(&Bs[buf][c * 8], bsrc + c * 8);
        }
    };

    const int kTiles = K >> 5;

    // prologue: tiles 0 and 1 in flight; drain tile 0 only
    load_tile(0, 0);
    if (kTiles > 1) load_tile(1, 1);
    wait_async4();
    __syncthreads();

    int cur = 0;
    for (int kt = 0; kt < kTiles; ++kt) {
        // batch all fragment loads for this k-step into registers
        v16h a = *(const v16h*)(&As[cur][(w * 32 + lane) << 4]);
        v16h bf[8];
        #pragma unroll
        for (int t = 0; t < 8; ++t)
            bf[t] = *(const v16h*)(&Bs[cur][(t << 9) + (lane << 4)]);

        // issue DMA for tile kt+2 (overwrites the buffer retired at kt-1)
        const bool issued = (kt + 2 < kTiles);
        if (issued) {
            int nxt = cur + 2; if (nxt >= 3) nxt -= 3;
            load_tile(kt + 2, nxt);
        }

        // 8 back-to-back WMMAs on distinct accumulators (no inter-WMMA hazard)
        #pragma unroll
        for (int t = 0; t < 8; ++t)
            acc[t] = __builtin_amdgcn_wmma_f32_16x16x32_f16(
                false, a, false, bf[t], (short)0, acc[t], false, false);

        // drain only tile kt+1 (issued a full iteration ago); keep kt+2 in flight
        if (issued) wait_async4();
        else        wait_async0();
        __syncthreads();

        ++cur; if (cur == 3) cur = 0;
    }

    // --- epilogue: C layout -> VGPR r holds M = r + 8*(lane>>4), N = lane&15
    const int mrow0 = mBase + w * 16 + ((lane >> 4) << 3);
    #pragma unroll
    for (int t = 0; t < 8; ++t) {
        int col = nBase + t * 16 + (lane & 15);
        float bvv = bias[col];
        #pragma unroll
        for (int r = 0; r < 8; ++r) {
            float x = acc[t][r] + bvv;
            if (do_gelu) x = gelu_exact_(x);
            size_t idx = (size_t)(mrow0 + r) * DDIM + col;
            if (res) x += res[idx];
            out[idx] = x;
        }
    }
}

// ---------------------------------------------------------------------------
// coord = mean over experts, emitted as f16 A-matrix (16384 x 512)
// ---------------------------------------------------------------------------
__global__ __launch_bounds__(256) void mean_kernel(const float* __restrict__ f,
                                                   _Float16* __restrict__ Aout) {
    const size_t tok = blockIdx.x;
    const int tid = threadIdx.x;
    const float* base = f + tok * (NE * DDIM);
    #pragma unroll
    for (int i = 0; i < 2; ++i) {
        int d = tid + i * 256;
        float s = 0.f;
        #pragma unroll
        for (int e = 0; e < NE; ++e) s += base[e * DDIM + d];
        Aout[tok * DDIM + d] = (_Float16)(s * 0.125f);
    }
}

// ---------------------------------------------------------------------------
// row LayerNorm over D=512
// ---------------------------------------------------------------------------
__global__ __launch_bounds__(256) void ln_kernel(const float* __restrict__ h,
                                                 const float* __restrict__ g,
                                                 const float* __restrict__ bb,
                                                 float* __restrict__ out) {
    __shared__ float red[256];
    __shared__ float s_mu, s_rstd;
    const size_t tok = blockIdx.x;
    const int tid = threadIdx.x;
    const float* row = h + tok * DDIM;
    float x0 = row[tid], x1 = row[tid + 256];

    red[tid] = x0 + x1;
    __syncthreads();
    for (int s = 128; s > 0; s >>= 1) {
        if (tid < s) red[tid] += red[tid + s];
        __syncthreads();
    }
    if (tid == 0) s_mu = red[0] * (1.0f / DDIM);
    __syncthreads();
    float mu = s_mu;
    float d0 = x0 - mu, d1 = x1 - mu;

    red[tid] = d0 * d0 + d1 * d1;
    __syncthreads();
    for (int s = 128; s > 0; s >>= 1) {
        if (tid < s) red[tid] += red[tid + s];
        __syncthreads();
    }
    if (tid == 0) s_rstd = rsqrtf(red[0] * (1.0f / DDIM) + 1e-5f);
    __syncthreads();
    float r = s_rstd;
    out[tok * DDIM + tid]       = d0 * r * g[tid]       + bb[tid];
    out[tok * DDIM + tid + 256] = d1 * r * g[tid + 256] + bb[tid + 256];
}

// ---------------------------------------------------------------------------
// launcher
// ---------------------------------------------------------------------------
extern "C" void kernel_launch(void* const* d_in, const int* in_sizes, int n_in,
                              void* d_out, int out_size, void* d_ws, size_t ws_size,
                              hipStream_t stream) {
    (void)in_sizes; (void)n_in; (void)out_size; (void)ws_size;

    const float* expert_outputs = (const float*)d_in[0];   // (8,2048,8,512)
    const float* adj_logits     = (const float*)d_in[1];   // (2,8,8)
    const float* w_msg          = (const float*)d_in[2];   // (2,1024)
    const float* b_msg          = (const float*)d_in[3];   // (2,)
    const float* Wn             = (const float*)d_in[4];   // (2,512,512)
    const float* bn             = (const float*)d_in[5];   // (2,512)
    const float* Ws             = (const float*)d_in[6];   // (2,512,512)
    const float* bs             = (const float*)d_in[7];   // (2,512)
    const float* Wc             = (const float*)d_in[8];   // (512,512)
    const float* bc             = (const float*)d_in[9];   // (512,)
    const float* ln_g           = (const float*)d_in[10];  // (512,)
    const float* ln_b           = (const float*)d_in[11];  // (512,)
    float* outp = (float*)d_out;                           // (8,2048,512)

    // ---- workspace layout ----
    char* wsb = (char*)d_ws;
    constexpr size_t SZ_F    = (size_t)NROWS * DDIM * sizeof(float);      // 268 MB
    constexpr size_t SZ_A    = (size_t)NROWS * 1024 * sizeof(_Float16);   // 268 MB
    constexpr size_t SZ_B    = (size_t)2 * 1024 * 512 * sizeof(_Float16); // 2 MB
    constexpr size_t SZ_BC   = (size_t)512 * 512 * sizeof(_Float16);      // 0.5 MB
    constexpr size_t SZ_BIAS = (size_t)2 * 512 * sizeof(float);
    float*    fbuf  = (float*)   (wsb);
    _Float16* Abuf  = (_Float16*)(wsb + SZ_F);
    _Float16* Bswz  = (_Float16*)(wsb + SZ_F + SZ_A);
    _Float16* Bcswz = (_Float16*)(wsb + SZ_F + SZ_A + SZ_B);
    float*    bias2 = (float*)   (wsb + SZ_F + SZ_A + SZ_B + SZ_BC);
    float*    hbuf  = (float*)   (wsb + SZ_F + SZ_A + SZ_B + SZ_BC + SZ_BIAS);

    // ---- weight preparation ----
    build_b_layer_kernel<<<dim3(2048, 2), 256, 0, stream>>>(Wn, Ws, Bswz);
    build_b_head_kernel<<<1024, 256, 0, stream>>>(Wc, Bcswz);
    build_bias_kernel<<<2, 512, 0, stream>>>(bn, bs, bias2);

    // ---- 2 GNN layers ----
    for (int l = 0; l < 2; ++l) {
        const float* fsrc = (l == 0) ? expert_outputs : fbuf;
        gate_msg_kernel<<<NTOK, 256, 0, stream>>>(
            fsrc, adj_logits + l * 64, w_msg + l * 1024, b_msg + l, Abuf);
        gemm_kernel<<<dim3(NROWS / 128, 4), 256, 0, stream>>>(
            Abuf, 1024, Bswz + (size_t)l * (1024 * 512), 1024,
            bias2 + l * 512, expert_outputs, fbuf, 1);
    }

    // ---- head: mean over experts -> GEMM+GELU -> LayerNorm ----
    mean_kernel<<<NTOK, 256, 0, stream>>>(fbuf, Abuf);
    gemm_kernel<<<dim3(NTOK / 128, 4), 256, 0, stream>>>(
        Abuf, 512, Bcswz, 512, bc, nullptr, hbuf, 1);
    ln_kernel<<<NTOK, 256, 0, stream>>>(hbuf, ln_g, ln_b, outp);
}
